// MonotonicNN_33380485824917
// MI455X (gfx1250) — matrix operand order repaired and checked by reading, not compiled, not run on hardware
//
#include <hip/hip_runtime.h>
#include <math.h>

typedef _Float16 v16h __attribute__((ext_vector_type(16)));
typedef _Float16 v8h  __attribute__((ext_vector_type(8)));
typedef float    v8f  __attribute__((ext_vector_type(8)));

#define HID    128
#define HDIM   64
#define NSTEP  50
#define SROW   144   // padded weight row (halfs): 288 B, every 8-half group 16B-aligned
#define C0ROW  80    // padded row for 64-wide cw0: 160 B
#define TROW   16    // transposed staging row (halfs) = 32 B

// A-fragment K order used by this code (contiguous; B loads use the same order):
// element e (0..15), half hi (0/1): k = 16*hi + e
__device__ __forceinline__ int kmapA(int e, int hi) {
    return (hi << 4) + e;
}

__device__ __forceinline__ v8h pack8(float4 u, float4 v) {
    v8h r;
    r[0] = (_Float16)u.x; r[1] = (_Float16)u.y; r[2] = (_Float16)u.z; r[3] = (_Float16)u.w;
    r[4] = (_Float16)v.x; r[5] = (_Float16)v.y; r[6] = (_Float16)v.z; r[7] = (_Float16)v.w;
    return r;
}

__device__ __forceinline__ void cvtA(v16h& a, float4 u0, float4 u1, float4 u2, float4 u3) {
    a[0]=(_Float16)u0.x; a[1]=(_Float16)u0.y; a[2]=(_Float16)u0.z; a[3]=(_Float16)u0.w;
    a[4]=(_Float16)u1.x; a[5]=(_Float16)u1.y; a[6]=(_Float16)u1.z; a[7]=(_Float16)u1.w;
    a[8]=(_Float16)u2.x; a[9]=(_Float16)u2.y; a[10]=(_Float16)u2.z; a[11]=(_Float16)u2.w;
    a[12]=(_Float16)u3.x; a[13]=(_Float16)u3.y; a[14]=(_Float16)u3.z; a[15]=(_Float16)u3.w;
}

// ---------------------------------------------------------------------------
// Clenshaw-Curtis weights + steps (padded to 64: ccw=0, steps=-1 -> xs=0)
// ---------------------------------------------------------------------------
__global__ void setup_tables(float* __restrict__ ccw, float* __restrict__ steps) {
    int j = threadIdx.x;
    if (j >= 64) return;
    const double pi = 3.14159265358979323846;
    if (j <= NSTEP) {
        steps[j] = (float)cos((double)j * pi / NSTEP);
        double s = 0.0;
        for (int i = 0; i <= NSTEP; ++i) {
            double w;
            if (i == 0)      w = 1.0;
            else if (i & 1)  w = 0.0;
            else             w = 2.0 / (1.0 - (double)i * (double)i);
            double l = cos((double)i * (double)j * pi / NSTEP);
            if (j == 0)     l = 0.5;
            if (j == NSTEP) l *= 0.5;
            l *= 2.0 / NSTEP;
            s += l * w;
        }
        ccw[j] = (float)s;
    } else {
        steps[j] = -1.0f;   // xs = x*(steps+1)/2 = 0 for padded rows
        ccw[j]   = 0.0f;
    }
}

// ---------------------------------------------------------------------------
// Cond net: offset[b] = relu(relu(relu(h@cw0^T+cb0)@cw1^T+cb1)@cw2^T+cb2) @ cw3[0] + cb3[0]
// One wave per 16-row tile. 16384 rows -> 1024 tiles -> 128 blocks x 8 waves.
// ---------------------------------------------------------------------------
__global__ __launch_bounds__(256) void cond_kernel(
    const float* __restrict__ h,
    const float* __restrict__ cw0, const float* __restrict__ cb0,
    const float* __restrict__ cw1, const float* __restrict__ cb1,
    const float* __restrict__ cw2, const float* __restrict__ cb2,
    const float* __restrict__ cw3, const float* __restrict__ cb3,
    float* __restrict__ offset_out)
{
    __shared__ alignas(32) _Float16 w0s[HID * C0ROW];
    __shared__ alignas(32) _Float16 w1s[HID * SROW];
    __shared__ alignas(32) _Float16 w2s[HID * SROW];
    __shared__ alignas(32) _Float16 stg[8][HID * TROW];   // [col][row] transposed
    __shared__ float b0s[HID], b1s[HID], b2s[HID], w3s[HID];

    const int tid = threadIdx.x;
    // vectorized weight fill: 8 f32 -> 8 f16 per step
    for (int i = tid; i < HID * HDIM / 8; i += 256) {      // 1024 groups
        int row = i >> 3, col8 = (i & 7) * 8;
        const float4* g = (const float4*)&cw0[row * HDIM + col8];
        *(v8h*)&w0s[row * C0ROW + col8] = pack8(g[0], g[1]);
    }
    for (int i = tid; i < HID * HID / 8; i += 256) {       // 2048 groups
        int row = i >> 4, col8 = (i & 15) * 8;
        const float4* g1 = (const float4*)&cw1[row * HID + col8];
        const float4* g2 = (const float4*)&cw2[row * HID + col8];
        *(v8h*)&w1s[row * SROW + col8] = pack8(g1[0], g1[1]);
        *(v8h*)&w2s[row * SROW + col8] = pack8(g2[0], g2[1]);
    }
    if (tid < HID) {
        b0s[tid] = cb0[tid]; b1s[tid] = cb1[tid];
        b2s[tid] = cb2[tid]; w3s[tid] = cw3[tid];   // row 0 of (2,128)
    }
    __syncthreads();

    const int wave = tid >> 5, lane = tid & 31;
    const int m = lane & 15, hi = lane >> 4;
    const int row0 = (blockIdx.x * 8 + wave) * 16;
    _Float16* st = &stg[wave][0];

    // ---- layer 0: A from h (K = 64), vectorized loads ----
    v16h a[4];
    {
        const float4* hp = (const float4*)&h[(size_t)(row0 + m) * HDIM];
        #pragma unroll
        for (int c = 0; c < 2; ++c) {
            float4 u0 = hp[8*c + 4*hi],     u1 = hp[8*c + 4*hi + 1];
            float4 u2 = hp[8*c + 4*hi + 2], u3 = hp[8*c + 4*hi + 3];
            cvtA(a[c], u0, u1, u2, u3);
        }
    }
    #pragma unroll
    for (int nt = 0; nt < 8; ++nt) {
        int n = nt * 16 + m;
        v8f acc; float bv = b0s[n];
        #pragma unroll
        for (int r = 0; r < 8; ++r) acc[r] = bv;
        #pragma unroll
        for (int c = 0; c < 2; ++c) {
            v16h bf = *(const v16h*)&w0s[n * C0ROW + 32 * c + 16 * hi];
            acc = __builtin_amdgcn_wmma_f32_16x16x32_f16(false, a[c], false, bf,
                                                         (short)0, acc, false, false);
        }
        v8h pk;
        #pragma unroll
        for (int r = 0; r < 8; ++r) pk[r] = (_Float16)fmaxf(acc[r], 0.0f);
        *(v8h*)&st[n * TROW + 8 * hi] = pk;     // rows 8hi..8hi+7 of column n
    }
    __syncthreads();
    #pragma unroll
    for (int c = 0; c < 4; ++c)
        #pragma unroll
        for (int e = 0; e < 16; ++e)
            a[c][e] = st[(32 * c + kmapA(e, hi)) * TROW + m];
    __syncthreads();

    // ---- layer 1 (K = 128) ----
    #pragma unroll
    for (int nt = 0; nt < 8; ++nt) {
        int n = nt * 16 + m;
        v8f acc; float bv = b1s[n];
        #pragma unroll
        for (int r = 0; r < 8; ++r) acc[r] = bv;
        #pragma unroll
        for (int c = 0; c < 4; ++c) {
            v16h bf = *(const v16h*)&w1s[n * SROW + 32 * c + 16 * hi];
            acc = __builtin_amdgcn_wmma_f32_16x16x32_f16(false, a[c], false, bf,
                                                         (short)0, acc, false, false);
        }
        v8h pk;
        #pragma unroll
        for (int r = 0; r < 8; ++r) pk[r] = (_Float16)fmaxf(acc[r], 0.0f);
        *(v8h*)&st[n * TROW + 8 * hi] = pk;
    }
    __syncthreads();
    #pragma unroll
    for (int c = 0; c < 4; ++c)
        #pragma unroll
        for (int e = 0; e < 16; ++e)
            a[c][e] = st[(32 * c + kmapA(e, hi)) * TROW + m];
    __syncthreads();

    // ---- layer 2 + dot with cw3 row 0 ----
    float p[8];
    #pragma unroll
    for (int r = 0; r < 8; ++r) p[r] = 0.0f;
    #pragma unroll
    for (int nt = 0; nt < 8; ++nt) {
        int n = nt * 16 + m;
        v8f acc; float bv = b2s[n];
        #pragma unroll
        for (int r = 0; r < 8; ++r) acc[r] = bv;
        #pragma unroll
        for (int c = 0; c < 4; ++c) {
            v16h bf = *(const v16h*)&w2s[n * SROW + 32 * c + 16 * hi];
            acc = __builtin_amdgcn_wmma_f32_16x16x32_f16(false, a[c], false, bf,
                                                         (short)0, acc, false, false);
        }
        float w3v = w3s[n];
        #pragma unroll
        for (int r = 0; r < 8; ++r) p[r] += fmaxf(acc[r], 0.0f) * w3v;
    }
    #pragma unroll
    for (int mask = 1; mask < 16; mask <<= 1)
        #pragma unroll
        for (int r = 0; r < 8; ++r)
            p[r] += __shfl_xor(p[r], mask, 32);

    if (m == 0) {
        float cb3v = cb3[0];
        #pragma unroll
        for (int r = 0; r < 8; ++r)
            offset_out[row0 + r + 8 * hi] = p[r] + cb3v;
    }
}

// ---------------------------------------------------------------------------
// Integrand: one wave per batch element b; 4 tiles of 16 quadrature steps.
// out[b] = (sum_s dz[b,s]*ccw[s]) * x[b] * 0.5 + offset[b]
// ---------------------------------------------------------------------------
__global__ __launch_bounds__(256) void integrand_kernel(
    const float* __restrict__ x,
    const float* __restrict__ iw0, const float* __restrict__ ib0,
    const float* __restrict__ iw1, const float* __restrict__ ib1,
    const float* __restrict__ iw2, const float* __restrict__ ib2,
    const float* __restrict__ iw3, const float* __restrict__ ib3,
    const float* __restrict__ ccw, const float* __restrict__ steps,
    const float* __restrict__ offset_in,
    float* __restrict__ out)
{
    __shared__ alignas(32) _Float16 w1s[HID * SROW];
    __shared__ alignas(32) _Float16 w2s[HID * SROW];
    __shared__ alignas(32) _Float16 stg[8][HID * TROW];   // [col][row] transposed
    __shared__ float w0c[HID], b0s[HID], b1s[HID], b2s[HID], w3s[HID];
    __shared__ float ccs[64], sts[64];

    const int tid = threadIdx.x;
    for (int i = tid; i < HID * HID / 8; i += 256) {       // 2048 groups of 8
        int row = i >> 4, col8 = (i & 15) * 8;
        const float4* g1 = (const float4*)&iw1[row * HID + col8];
        const float4* g2 = (const float4*)&iw2[row * HID + col8];
        *(v8h*)&w1s[row * SROW + col8] = pack8(g1[0], g1[1]);
        *(v8h*)&w2s[row * SROW + col8] = pack8(g2[0], g2[1]);
    }
    if (tid < HID) {
        w0c[tid] = iw0[tid * 2];      // column 0 of (128,2); z[...,1]==0
        b0s[tid] = ib0[tid];
        b1s[tid] = ib1[tid]; b2s[tid] = ib2[tid]; w3s[tid] = iw3[tid];
    }
    if (tid < 64) { ccs[tid] = ccw[tid]; sts[tid] = steps[tid]; }
    __syncthreads();

    const int wave = tid >> 5, lane = tid & 31;
    const int m = lane & 15, hi = lane >> 4;
    const int b = blockIdx.x * 8 + wave;
    const float xv   = x[b];
    const float ib3v = ib3[0];
    _Float16* st = &stg[wave][0];
    float zacc = 0.0f;

    for (int t = 0; t < 4; ++t) {
        // ---- analytic layer 0: a0[k] = relu(iw0[k,0]*xs + ib0[k]) ----
        const int srow = t * 16 + m;
        const float xs = xv * (sts[srow] + 1.0f) * 0.5f;
        v16h a[4];
        #pragma unroll
        for (int c = 0; c < 4; ++c)
            #pragma unroll
            for (int e = 0; e < 16; ++e) {
                int k = 32 * c + kmapA(e, hi);
                a[c][e] = (_Float16)fmaxf(w0c[k] * xs + b0s[k], 0.0f);
            }

        // ---- layer 1 (K = 128) ----
        #pragma unroll
        for (int nt = 0; nt < 8; ++nt) {
            int n = nt * 16 + m;
            v8f acc; float bv = b1s[n];
            #pragma unroll
            for (int r = 0; r < 8; ++r) acc[r] = bv;
            #pragma unroll
            for (int c = 0; c < 4; ++c) {
                v16h bf = *(const v16h*)&w1s[n * SROW + 32 * c + 16 * hi];
                acc = __builtin_amdgcn_wmma_f32_16x16x32_f16(false, a[c], false, bf,
                                                             (short)0, acc, false, false);
            }
            v8h pk;
            #pragma unroll
            for (int r = 0; r < 8; ++r) pk[r] = (_Float16)fmaxf(acc[r], 0.0f);
            *(v8h*)&st[n * TROW + 8 * hi] = pk;   // one 16B store per column tile
        }
        __syncthreads();
        #pragma unroll
        for (int c = 0; c < 4; ++c)
            #pragma unroll
            for (int e = 0; e < 16; ++e)
                a[c][e] = st[(32 * c + kmapA(e, hi)) * TROW + m];
        __syncthreads();

        // ---- layer 2 + elu-dot with iw3 ----
        float p[8];
        #pragma unroll
        for (int r = 0; r < 8; ++r) p[r] = 0.0f;
        #pragma unroll
        for (int nt = 0; nt < 8; ++nt) {
            int n = nt * 16 + m;
            v8f acc; float bv = b2s[n];
            #pragma unroll
            for (int r = 0; r < 8; ++r) acc[r] = bv;
            #pragma unroll
            for (int c = 0; c < 4; ++c) {
                v16h bf = *(const v16h*)&w2s[n * SROW + 32 * c + 16 * hi];
                acc = __builtin_amdgcn_wmma_f32_16x16x32_f16(false, a[c], false, bf,
                                                             (short)0, acc, false, false);
            }
            float w3v = w3s[n];
            #pragma unroll
            for (int r = 0; r < 8; ++r) p[r] += fmaxf(acc[r], 0.0f) * w3v;
        }
        #pragma unroll
        for (int mask = 1; mask < 16; mask <<= 1)
            #pragma unroll
            for (int r = 0; r < 8; ++r)
                p[r] += __shfl_xor(p[r], mask, 32);

        // dz = elu(v)+1+1e-8 ; weighted CC sum over this tile's 16 steps
        float tsum = 0.0f;
        #pragma unroll
        for (int r = 0; r < 8; ++r) {
            int s = t * 16 + r + 8 * hi;
            float v  = p[r] + ib3v;
            float dz = (v > 0.0f) ? (v + 1.0f + 1e-8f) : (expf(v) + 1e-8f);
            tsum += dz * ccs[s];
        }
        tsum += __shfl_xor(tsum, 16, 32);  // combine the two row-halves
        zacc += tsum;
    }

    if (lane == 0)
        out[b] = zacc * xv * 0.5f + offset_in[b];
}

// ---------------------------------------------------------------------------
extern "C" void kernel_launch(void* const* d_in, const int* in_sizes, int n_in,
                              void* d_out, int out_size, void* d_ws, size_t ws_size,
                              hipStream_t stream) {
    (void)in_sizes; (void)n_in; (void)out_size; (void)ws_size;
    const float* x   = (const float*)d_in[0];
    const float* h   = (const float*)d_in[1];
    const float* iw0 = (const float*)d_in[2];
    const float* ib0 = (const float*)d_in[3];
    const float* iw1 = (const float*)d_in[4];
    const float* ib1 = (const float*)d_in[5];
    const float* iw2 = (const float*)d_in[6];
    const float* ib2 = (const float*)d_in[7];
    const float* iw3 = (const float*)d_in[8];
    const float* ib3 = (const float*)d_in[9];
    const float* cw0 = (const float*)d_in[10];
    const float* cb0 = (const float*)d_in[11];
    const float* cw1 = (const float*)d_in[12];
    const float* cb1 = (const float*)d_in[13];
    const float* cw2 = (const float*)d_in[14];
    const float* cb2 = (const float*)d_in[15];
    const float* cw3 = (const float*)d_in[16];
    const float* cb3 = (const float*)d_in[17];

    float* ws     = (float*)d_ws;
    float* ccw    = ws;        // 64 floats
    float* steps  = ws + 64;   // 64 floats
    float* offset = ws + 128;  // 16384 floats

    setup_tables<<<1, 64, 0, stream>>>(ccw, steps);
    cond_kernel<<<128, 256, 0, stream>>>(h, cw0, cb0, cw1, cb1, cw2, cb2,
                                         cw3, cb3, offset);
    integrand_kernel<<<2048, 256, 0, stream>>>(x, iw0, ib0, iw1, ib1, iw2, ib2,
                                               iw3, ib3, ccw, steps, offset,
                                               (float*)d_out);
}
